// LSTMCustom_32332513804757
// MI455X (gfx1250) — matrix operand order repaired and checked by reading, not compiled, not run on hardware
//
#include <hip/hip_runtime.h>
#include <hip/hip_bf16.h>
#include <math.h>

// ---------------------------------------------------------------------------
// 2-layer LSTM for MI455X (gfx1250, wave32, WMMA).
//   B=64, S=512, CIN=1024, H=1024
// Strategy:
//   * Weights fp32 -> bf16, transposed to [K=2048, N=4096]; both matrices are
//     32 MB bf16 total, resident in the 192 MB L2 across the 512-step scan.
//   * Per step, per cell: z[64,4096] = [x_t|h] @ Wt via v_wmma_f32_16x16x32_bf16.
//     A panel is DMA'd into double-buffered LDS with global_load_async_to_lds_b128
//     (ASYNCcnt), overlapping the next chunk's staging with current WMMA work.
//     B tiles stream straight from L2 as single-bf16 loads, row stride Nsz.
//   * Gates kernel: bias + sigmoid/tanh, updates c, writes h (bf16 + f32).
//     Layer-1 gates write h directly into d_out each step.
// ---------------------------------------------------------------------------

typedef __bf16 bf16;
typedef __attribute__((ext_vector_type(16))) __bf16 bf16x16;
typedef __attribute__((ext_vector_type(8)))  float  floatx8;

#define Bsz   64
#define Ssz   512
#define Hsz   1024
#define Ksz   2048      // CIN + H  (== H + H)
#define Nsz   4096      // 4*H
#define KCHUNK 128
#define NCH   (Ksz / KCHUNK)   // 16 chunks
#define APAD  8         // pad LDS rows by 8 bf16 (16B) -> conflict-free ds_load_b128

// ---------------- weight transpose + fp32->bf16 ----------------------------
// W[N, K] row-major fp32  ->  Wt[K, N] row-major bf16
__global__ void transpose_convert_kernel(const float* __restrict__ W,
                                         bf16* __restrict__ Wt) {
  __shared__ float tile[32][33];
  const int kb = blockIdx.x * 32;
  const int nb = blockIdx.y * 32;
  const int tx = threadIdx.x;        // 0..31
  const int ty = threadIdx.y;        // 0..7
  for (int i = ty; i < 32; i += 8)
    tile[i][tx] = W[(size_t)(nb + i) * Ksz + kb + tx];
  __syncthreads();
  for (int i = ty; i < 32; i += 8)
    Wt[(size_t)(kb + i) * Nsz + nb + tx] = (bf16)tile[tx][i];
}

// ---------------- bulk fp32 -> bf16 (for x) --------------------------------
__global__ void convert_bf16_kernel(const float* __restrict__ in,
                                    bf16* __restrict__ out, size_t n) {
  size_t i = ((size_t)blockIdx.x * blockDim.x + threadIdx.x) * 4;
  const size_t stride = (size_t)gridDim.x * blockDim.x * 4;
  for (; i < n; i += stride) {
    float4 v = *(const float4*)(in + i);
    out[i + 0] = (bf16)v.x;
    out[i + 1] = (bf16)v.y;
    out[i + 2] = (bf16)v.z;
    out[i + 3] = (bf16)v.w;
  }
}

// ---------------- zero-init recurrent state --------------------------------
__global__ void init_state_kernel(float* c0, float* c1, bf16* h0b, bf16* h1b) {
  const int i = blockIdx.x * blockDim.x + threadIdx.x;   // 0 .. 65535
  c0[i] = 0.0f;
  c1[i] = 0.0f;
  h0b[i] = (bf16)0.0f;
  h1b[i] = (bf16)0.0f;
}

// ---------------- WMMA GEMM: Z[64,4096] = [Alo|Ahi] @ Wt -------------------
// A is [64, 2048] bf16: first 1024 K from Alo (ld lda_lo), rest from Ahi.
// Block: 128 threads = 4 waves; each wave owns a 64x16 (MxN) slab (4 accums).
// A panel double-buffered in LDS via async DMA (global_load_async_to_lds_b128).
// Grid: Nsz/64 = 64 blocks.
__global__ __launch_bounds__(128)
void lstm_gemm_kernel(const bf16* __restrict__ Alo, int lda_lo,
                      const bf16* __restrict__ Ahi, int lda_hi,
                      const bf16* __restrict__ Wt,   // [Ksz, Nsz] bf16
                      float* __restrict__ Z) {       // [Bsz, Nsz] fp32
  __shared__ bf16 As[2][Bsz][KCHUNK + APAD];   // 2 x 17.4 KB

  const int tid  = threadIdx.x;
  const int wave = tid >> 5;
  const int lane = tid & 31;
  const int n0   = blockIdx.x * 64 + wave * 16;

  // Issue the async DMA for chunk `c` into LDS buffer `buf`:
  // 64 rows x 8 b128 (8 bf16 each, at 16-bf16 steps) = 512 x b128 transfers, 4 per thread.
  auto stage = [&](int c, int buf) {
    const int  kc  = c * KCHUNK;
    const bf16* src = (kc < Hsz) ? (Alo + kc) : (Ahi + (kc - Hsz));
    const int   ld  = (kc < Hsz) ? lda_lo : lda_hi;
#pragma unroll
    for (int j = 0; j < 4; ++j) {
      const int idx = j * 128 + tid;       // 0..511
      const int r   = idx >> 3;            // 0..63
      const int cc  = (idx & 7) << 4;      // 0,16,...,112
      const bf16* g = src + (size_t)r * ld + cc;
      // Low 32 bits of a generic LDS pointer == wave-relative LDS offset.
      const unsigned lofs = (unsigned)(uintptr_t)&As[buf][r][cc];
      asm volatile("global_load_async_to_lds_b128 %0, %1, off"
                   :: "v"(lofs), "v"(g)
                   : "memory");
    }
  };

  floatx8 acc[4] = {};   // 4 M-tiles (M = 0,16,32,48) x one 16-wide N tile

  stage(0, 0);
  for (int c = 0; c < NCH; ++c) {
    // Our LDS reads of the buffer the next DMA will overwrite must be done
    // (s_wait_dscnt 0), and our own DMA writes must have landed
    // (s_wait_asynccnt 0), before we pass the workgroup barrier.
    asm volatile("s_wait_dscnt 0\n\t"
                 "s_wait_asynccnt 0" ::: "memory");
    __syncthreads();
    if (c + 1 < NCH) stage(c + 1, (c + 1) & 1);   // overlap DMA with WMMA below

    const int buf = c & 1;
    const int kc  = c * KCHUNK;
#pragma unroll
    for (int kk = 0; kk < KCHUNK; kk += 32) {
      // B fragment (32x16): lane%16 = N column; lanes 0-15 hold K {0-7,16-23},
      // lanes 16-31 hold K {8-15,24-31} from Wt[K, N].
      union { bf16x16 v; uint4 u[2]; } bfrag;
      {
        const bf16* bp = Wt + (size_t)(kc + kk + (lane >> 4) * 8) * Nsz + n0 + (lane & 15);
#pragma unroll
        for (int s = 0; s < 8; ++s) {
          bfrag.v[s]     = bp[(size_t)s * Nsz];
          bfrag.v[8 + s] = bp[(size_t)(16 + s) * Nsz];
        }
      }
      const int m_lane = lane & 15;
      const int khalf  = (lane >> 4) * 8;  // lanes 0-15: K octets {0-7,16-23}; 16-31: {8-15,24-31}
#pragma unroll
      for (int mt = 0; mt < 4; ++mt) {
        union { bf16x16 v; uint4 u[2]; } afrag;
        const int m = mt * 16 + m_lane;
        afrag.u[0] = *(const uint4*)&As[buf][m][kk + khalf];
        afrag.u[1] = *(const uint4*)&As[buf][m][kk + 16 + khalf];
        acc[mt] = __builtin_amdgcn_wmma_f32_16x16x32_bf16(
            false, afrag.v, false, bfrag.v, (short)0, acc[mt], false, false);
      }
    }
  }

  // ---- writeback: C/D layout -> VGPR r holds M=r (lanes 0-15) / M=8+r ----
  const int nw    = n0 + (lane & 15);
  const int mbase = (lane >> 4) * 8;
#pragma unroll
  for (int mt = 0; mt < 4; ++mt) {
#pragma unroll
    for (int r = 0; r < 8; ++r) {
      Z[(size_t)(mt * 16 + mbase + r) * Nsz + nw] = acc[mt][r];
    }
  }
}

// ---------------- gates: bias + nonlinearities + state update --------------
__device__ __forceinline__ float sigf(float x) {
  return 1.0f / (1.0f + __expf(-x));
}

__global__ void lstm_gates_kernel(const float* __restrict__ Z,     // [64, 4096]
                                  const float* __restrict__ bias,  // [4096]
                                  float* __restrict__ c,           // [64, 1024]
                                  bf16* __restrict__ h_bf16,       // [64, 1024]
                                  float* __restrict__ h_f32) {     // or nullptr
  const int idx = blockIdx.x * blockDim.x + threadIdx.x;  // 0..65535
  const int b = idx >> 10;
  const int j = idx & 1023;
  const float* zr = Z + (size_t)b * Nsz;
  const float f = sigf (zr[j]            + bias[j]);
  const float i = sigf (zr[Hsz + j]      + bias[Hsz + j]);
  const float g = tanhf(zr[2 * Hsz + j]  + bias[2 * Hsz + j]);
  const float o = sigf (zr[3 * Hsz + j]  + bias[3 * Hsz + j]);
  const float cn = f * c[idx] + i * g;
  c[idx] = cn;
  const float hn = o * tanhf(cn);
  h_bf16[idx] = (bf16)hn;
  if (h_f32) h_f32[idx] = hn;
}

// ---------------------------------------------------------------------------
extern "C" void kernel_launch(void* const* d_in, const int* in_sizes, int n_in,
                              void* d_out, int out_size, void* d_ws, size_t ws_size,
                              hipStream_t stream) {
  const float* x  = (const float*)d_in[0];  // [64, 512, 1024]
  const float* W0 = (const float*)d_in[1];  // [4096, 2048]
  const float* b0 = (const float*)d_in[2];  // [4096]
  const float* W1 = (const float*)d_in[3];  // [4096, 2048]
  const float* b1 = (const float*)d_in[4];  // [4096]
  float* out = (float*)d_out;               // [64, 1024]

  // ---- carve workspace (256B aligned) ----
  uintptr_t p = (uintptr_t)d_ws;
  auto carve = [&](size_t bytes) -> void* {
    p = (p + 255) & ~(uintptr_t)255;
    void* r = (void*)p;
    p += bytes;
    return r;
  };
  bf16*  Wt0 = (bf16*) carve((size_t)Ksz * Nsz * sizeof(bf16));       // 16 MB
  bf16*  Wt1 = (bf16*) carve((size_t)Ksz * Nsz * sizeof(bf16));       // 16 MB
  bf16*  xb  = (bf16*) carve((size_t)Bsz * Ssz * Hsz * sizeof(bf16)); // 64 MB
  bf16*  h0b = (bf16*) carve((size_t)Bsz * Hsz * sizeof(bf16));
  bf16*  h1b = (bf16*) carve((size_t)Bsz * Hsz * sizeof(bf16));
  float* c0  = (float*)carve((size_t)Bsz * Hsz * sizeof(float));
  float* c1  = (float*)carve((size_t)Bsz * Hsz * sizeof(float));
  float* z0  = (float*)carve((size_t)Bsz * Nsz * sizeof(float));
  float* z1  = (float*)carve((size_t)Bsz * Nsz * sizeof(float));
  (void)ws_size; (void)in_sizes; (void)n_in; (void)out_size;

  // ---- one-time (per call) transforms: weights -> bf16 [K,N], x -> bf16 ----
  {
    dim3 blk(32, 8);
    dim3 grd(Ksz / 32, Nsz / 32);
    transpose_convert_kernel<<<grd, blk, 0, stream>>>(W0, Wt0);
    transpose_convert_kernel<<<grd, blk, 0, stream>>>(W1, Wt1);
  }
  {
    const size_t nx = (size_t)Bsz * Ssz * Hsz;   // 33554432, /4 per thread
    convert_bf16_kernel<<<8192, 256, 0, stream>>>(x, xb, nx);
  }
  init_state_kernel<<<(Bsz * Hsz) / 256, 256, 0, stream>>>(c0, c1, h0b, h1b);

  // ---- recurrence: 512 steps x (gemm + gates) x 2 layers ----
  const int gemm_blocks = Nsz / 64;   // 64 blocks, 128 threads each
  for (int s = 0; s < Ssz; ++s) {
    // layer 0: A = [x_t | h0]
    lstm_gemm_kernel<<<gemm_blocks, 128, 0, stream>>>(
        xb + (size_t)s * Hsz, Ssz * Hsz,   // x row stride = S*CIN (layout [B,S,CIN])
        h0b, Hsz, Wt0, z0);
    lstm_gates_kernel<<<(Bsz * Hsz) / 256, 256, 0, stream>>>(
        z0, b0, c0, h0b, nullptr);
    // layer 1: A = [h0 | h1]
    lstm_gemm_kernel<<<gemm_blocks, 128, 0, stream>>>(
        h0b, Hsz, h1b, Hsz, Wt1, z1);
    lstm_gates_kernel<<<(Bsz * Hsz) / 256, 256, 0, stream>>>(
        z1, b1, c1, h1b, out);             // h1 (f32) lands in d_out every step
  }
}
